// MultiHeadSelfAttention_18279380811924
// MI455X (gfx1250) — matrix operand order repaired and checked by reading, not compiled, not run on hardware
//
#include <hip/hip_runtime.h>

// ---------------------------------------------------------------------------
// CDNA5 (gfx1250) multi-head self-attention, bf16 WMMA pipeline.
// B=4, N=2048, D=1024, HEADS=16, HEAD_DIM=64.
// ---------------------------------------------------------------------------

typedef __attribute__((ext_vector_type(16))) __bf16 bf16x16;
typedef __attribute__((ext_vector_type(8)))  float  f32x8;
typedef __attribute__((ext_vector_type(4)))  int    v4i;

union Frag {
    bf16x16 v;
    uint4   q[2];
    __bf16  e[16];
};

__device__ __forceinline__ f32x8 zero8() {
    f32x8 z = {0.f, 0.f, 0.f, 0.f, 0.f, 0.f, 0.f, 0.f};
    return z;
}

__device__ __forceinline__ unsigned int pack_bf(float a, float b) {
    unsigned short lo = __builtin_bit_cast(unsigned short, (__bf16)a);
    unsigned short hi = __builtin_bit_cast(unsigned short, (__bf16)b);
    return (unsigned int)lo | ((unsigned int)hi << 16);
}

// ---------------------------------------------------------------------------
// Async global->LDS staging (GLOBAL_LOAD_ASYNC_TO_LDS_B128, ASYNCcnt).
// Builtin takes (addrspace(1) v4i* gsrc, addrspace(3) v4i* lds, imm off, imm cpol).
// ---------------------------------------------------------------------------
#if __has_builtin(__builtin_amdgcn_global_load_async_to_lds_b128)
#define USE_ASYNC_LDS 1
typedef __attribute__((address_space(1))) v4i gv4i;
typedef __attribute__((address_space(3))) v4i lv4i;

__device__ __forceinline__ void async_copy16(void* lds_dst, const void* gsrc) {
    __builtin_amdgcn_global_load_async_to_lds_b128((gv4i*)gsrc, (lv4i*)lds_dst,
                                                   /*offset=*/0, /*cpol=*/0);
}
__device__ __forceinline__ void async_wait0() {
#if __has_builtin(__builtin_amdgcn_s_wait_asynccnt)
    __builtin_amdgcn_s_wait_asynccnt(0);
#else
    asm volatile("s_wait_asynccnt 0" ::: "memory");
#endif
}
#endif

// A fragment (16x32 bf16, M x K): lane m = lane&15, half picks K sub-ranges
// {h*8..h*8+7} and {16+h*8..+7} -> two contiguous 16B loads.
__device__ __forceinline__ void load_fragA(Frag& f, const __bf16* base, int ld) {
    int lane = threadIdx.x & 31;
    const __bf16* p = base + (size_t)(lane & 15) * ld + ((lane >> 4) << 3);
    f.q[0] = *(const uint4*)p;
    f.q[1] = *(const uint4*)(p + 16);
}

// B fragment (32x16 bf16, K x N): lane n = lane&15, half picks K rows
// {h*8..h*8+7} and {16+h*8..+7} (strided loads down column n).
__device__ __forceinline__ void load_fragB(Frag& f, const __bf16* base, int ld) {
    int lane = threadIdx.x & 31;
    const __bf16* p = base + (lane & 15) + (size_t)((lane >> 4) << 3) * ld;
#pragma unroll
    for (int s = 0; s < 8; s++) {
        f.e[s]     = p[(size_t)s * ld];
        f.e[8 + s] = p[(size_t)(16 + s) * ld];
    }
}

__device__ __forceinline__ f32x8 wmma_bf16(const Frag& a, const Frag& b, f32x8 c) {
    return __builtin_amdgcn_wmma_f32_16x16x32_bf16(
        /*neg_a=*/false, a.v, /*neg_b=*/false, b.v,
        /*c_mod=*/(short)0, c, /*reuse_a=*/false, /*reuse_b=*/false);
}

// ---------------------------------------------------------------------------
// fp32 -> bf16 conversion, 4 elements/thread.
// ---------------------------------------------------------------------------
__global__ void __launch_bounds__(256) cvt_kernel(const float* __restrict__ in,
                                                  __bf16* __restrict__ out, int n4) {
    int i = blockIdx.x * 256 + threadIdx.x;
    if (i < n4) {
        float4 f = ((const float4*)in)[i];
        uint2 p;
        p.x = pack_bf(f.x, f.y);
        p.y = pack_bf(f.z, f.w);
        ((uint2*)out)[i] = p;
    }
}

// ---------------------------------------------------------------------------
// WMMA GEMM: C[M x N] = A[M x K] * B[K x N], A/B bf16 row-major.
// Each wave computes a 64x32 tile (4x2 of 16x16), 8 WMMAs per K=32 step.
// mode 0: bf16 row-major out; mode 1: bf16 transposed out (Ct[n][m], ldct);
// mode 2: f32 out + bias.
// ---------------------------------------------------------------------------
__global__ void __launch_bounds__(128) gemm_kernel(
    const __bf16* __restrict__ A, const __bf16* __restrict__ B,
    void* __restrict__ C, const float* __restrict__ bias,
    int M, int N, int Kd, int mode, int ldct) {

    int wave   = (blockIdx.x * 128 + threadIdx.x) >> 5;
    int tilesN = N >> 5;
    int wm = wave / tilesN;
    int wn = wave - wm * tilesN;
    int row0 = wm << 6;
    int col0 = wn << 5;

    f32x8 c[4][2];
#pragma unroll
    for (int i = 0; i < 4; i++)
#pragma unroll
        for (int j = 0; j < 2; j++) c[i][j] = zero8();

    for (int k0 = 0; k0 < Kd; k0 += 32) {
        Frag a[4], b[2];
#pragma unroll
        for (int i = 0; i < 4; i++)
            load_fragA(a[i], A + (size_t)(row0 + i * 16) * Kd + k0, Kd);
#pragma unroll
        for (int j = 0; j < 2; j++)
            load_fragB(b[j], B + (size_t)k0 * N + col0 + j * 16, N);
#pragma unroll
        for (int i = 0; i < 4; i++)
#pragma unroll
            for (int j = 0; j < 2; j++)
                c[i][j] = wmma_bf16(a[i], b[j], c[i][j]);
    }

    int lane = threadIdx.x & 31;
    int cn = lane & 15, hf = lane >> 4;

    if (mode == 2) {
        float* O = (float*)C;
#pragma unroll
        for (int i = 0; i < 4; i++) {
#pragma unroll
            for (int j = 0; j < 2; j++) {
                int col = col0 + j * 16 + cn;
                float bb = bias[col];
#pragma unroll
                for (int r = 0; r < 8; r++)
                    O[(size_t)(row0 + i * 16 + hf * 8 + r) * N + col] = c[i][j][r] + bb;
            }
        }
    } else if (mode == 1) {
        // transposed bf16 store: Ct[col][row]; 8 consecutive tokens per lane -> b128.
        __bf16* O = (__bf16*)C;
#pragma unroll
        for (int i = 0; i < 4; i++) {
#pragma unroll
            for (int j = 0; j < 2; j++) {
                int col = col0 + j * 16 + cn;
                int rb  = row0 + i * 16 + hf * 8;
                uint4 pk;
                pk.x = pack_bf(c[i][j][0], c[i][j][1]);
                pk.y = pack_bf(c[i][j][2], c[i][j][3]);
                pk.z = pack_bf(c[i][j][4], c[i][j][5]);
                pk.w = pack_bf(c[i][j][6], c[i][j][7]);
                *(uint4*)(O + (size_t)col * ldct + rb) = pk;
            }
        }
    } else {
        __bf16* O = (__bf16*)C;
#pragma unroll
        for (int i = 0; i < 4; i++) {
#pragma unroll
            for (int j = 0; j < 2; j++) {
                int col = col0 + j * 16 + cn;
#pragma unroll
                for (int r = 0; r < 8; r++)
                    O[(size_t)(row0 + i * 16 + hf * 8 + r) * N + col] = (__bf16)c[i][j][r];
            }
        }
    }
}

// ---------------------------------------------------------------------------
// Flash attention: grid (32 q-blocks, 64 b*h), 128 threads = 4 waves.
// Each wave owns a 16-row query tile; the block cooperatively stages
// Kt (64 x 32) and V (32 x 64) chunks in LDS via async global->LDS copies
// (padded pitches, conflict-free). Online softmax with shfl_xor over the
// 16-lane row groups.
// ---------------------------------------------------------------------------
__global__ void __launch_bounds__(128) attn_kernel(
    const __bf16* __restrict__ Q,   // [8192][1024] token-major
    const __bf16* __restrict__ Kt,  // [1024][8192] feature-major (transposed)
    const __bf16* __restrict__ V,   // [8192][1024] token-major
    __bf16* __restrict__ ctx) {     // [8192][1024]

    const int PK = 40;  // KsT pitch (bf16): 64 d-rows x 32 keys
    const int PV = 72;  // Vs pitch  (bf16): 32 key-rows x 64 d
    const int PP = 36;  // Ps pitch  (f32) : 16 rows x 32 keys
    __shared__ __bf16 KsT[64 * 40];
    __shared__ __bf16 Vs[32 * 72];
    __shared__ float  Ps[4][16 * 36];

    int b = blockIdx.y >> 4, h = blockIdx.y & 15;
    int wave = threadIdx.x >> 5, lane = threadIdx.x & 31;
    int q0 = blockIdx.x * 64 + wave * 16;

    Frag aq0, aq1;
    const __bf16* qb = Q + (size_t)(b * 2048 + q0) * 1024 + h * 64;
    load_fragA(aq0, qb, 1024);       // K-dim 0..31
    load_fragA(aq1, qb + 32, 1024);  // K-dim 32..63

    f32x8 o[4];
#pragma unroll
    for (int dt = 0; dt < 4; dt++) o[dt] = zero8();
    float mrow[8], lrow[8];
#pragma unroll
    for (int r = 0; r < 8; r++) { mrow[r] = -3.0e38f; lrow[r] = 0.f; }

    const __bf16* Kg = Kt + (size_t)(h * 64) * 8192 + (size_t)b * 2048;
    const __bf16* Vg = V + (size_t)(b * 2048) * 1024 + h * 64;

    for (int kb = 0; kb < 2048; kb += 32) {
        __syncthreads();
        {
            int t = threadIdx.x;
#ifdef USE_ASYNC_LDS
            // KsT: 64 rows x 32 keys = 256 x b128; Vs: 32 rows x 64 d = 256 x b128
#pragma unroll
            for (int i = 0; i < 2; i++) {
                int idx = t + i * 128;
                int d = idx >> 2, qd = idx & 3;
                async_copy16((uint4*)(KsT + d * PK) + qd,
                             (const uint4*)(Kg + (size_t)d * 8192 + kb) + qd);
            }
#pragma unroll
            for (int i = 0; i < 2; i++) {
                int idx = t + i * 128;
                int kr = idx >> 3, qd = idx & 7;
                async_copy16((uint4*)(Vs + kr * PV) + qd,
                             (const uint4*)(Vg + (size_t)(kb + kr) * 1024) + qd);
            }
            async_wait0();
#else
#pragma unroll
            for (int i = 0; i < 2; i++) {
                int idx = t + i * 128;
                int d = idx >> 2, qd = idx & 3;
                *((uint4*)(KsT + d * PK) + qd) =
                    *((const uint4*)(Kg + (size_t)d * 8192 + kb) + qd);
            }
#pragma unroll
            for (int i = 0; i < 2; i++) {
                int idx = t + i * 128;
                int kr = idx >> 3, qd = idx & 7;
                *((uint4*)(Vs + kr * PV) + qd) =
                    *((const uint4*)(Vg + (size_t)(kb + kr) * 1024) + qd);
            }
#endif
        }
        __syncthreads();

        // S = Q * K^T for 32 keys: 2 N-subtiles x 2 K-slices = 4 WMMAs.
        f32x8 s[2];
#pragma unroll
        for (int kt = 0; kt < 2; kt++) {
            f32x8 acc = zero8();
            Frag bk0, bk1;
            load_fragB(bk0, KsT + kt * 16, PK);            // d 0..31
            load_fragB(bk1, KsT + 32 * PK + kt * 16, PK);  // d 32..63
            acc = wmma_bf16(aq0, bk0, acc);
            acc = wmma_bf16(aq1, bk1, acc);
#pragma unroll
            for (int r = 0; r < 8; r++) s[kt][r] = acc[r] * 0.125f;  // 1/sqrt(64)
        }

        // online softmax (rows map to vgpr r + 8*half in both S and O tiles)
        float alpha[8];
#pragma unroll
        for (int r = 0; r < 8; r++) {
            float mx = fmaxf(s[0][r], s[1][r]);
            mx = fmaxf(mx, __shfl_xor(mx, 1, 32));
            mx = fmaxf(mx, __shfl_xor(mx, 2, 32));
            mx = fmaxf(mx, __shfl_xor(mx, 4, 32));
            mx = fmaxf(mx, __shfl_xor(mx, 8, 32));
            float mnew = fmaxf(mrow[r], mx);
            alpha[r] = __expf(mrow[r] - mnew);
            mrow[r] = mnew;
            float p0 = __expf(s[0][r] - mnew);
            float p1 = __expf(s[1][r] - mnew);
            s[0][r] = p0;
            s[1][r] = p1;
            float sum = p0 + p1;
            sum += __shfl_xor(sum, 1, 32);
            sum += __shfl_xor(sum, 2, 32);
            sum += __shfl_xor(sum, 4, 32);
            sum += __shfl_xor(sum, 8, 32);
            lrow[r] = lrow[r] * alpha[r] + sum;
        }
#pragma unroll
        for (int dt = 0; dt < 4; dt++)
#pragma unroll
            for (int r = 0; r < 8; r++) o[dt][r] *= alpha[r];

        // P: C-layout -> A-frag layout via per-wave LDS (no barrier: same wave).
        float* ps = Ps[wave];
        {
            int cn = lane & 15, hf = lane >> 4;
#pragma unroll
            for (int r = 0; r < 8; r++) {
                ps[(r + hf * 8) * PP + cn]      = s[0][r];
                ps[(r + hf * 8) * PP + 16 + cn] = s[1][r];
            }
        }
        Frag ap;
        {
            int m = lane & 15, hf = lane >> 4;
            const float* pr = ps + m * PP + hf * 8;
            float4 t0 = *(const float4*)pr;
            float4 t1 = *(const float4*)(pr + 4);
            float4 t2 = *(const float4*)(pr + 16);
            float4 t3 = *(const float4*)(pr + 20);
            ap.e[0]  = (__bf16)t0.x; ap.e[1]  = (__bf16)t0.y;
            ap.e[2]  = (__bf16)t0.z; ap.e[3]  = (__bf16)t0.w;
            ap.e[4]  = (__bf16)t1.x; ap.e[5]  = (__bf16)t1.y;
            ap.e[6]  = (__bf16)t1.z; ap.e[7]  = (__bf16)t1.w;
            ap.e[8]  = (__bf16)t2.x; ap.e[9]  = (__bf16)t2.y;
            ap.e[10] = (__bf16)t2.z; ap.e[11] = (__bf16)t2.w;
            ap.e[12] = (__bf16)t3.x; ap.e[13] = (__bf16)t3.y;
            ap.e[14] = (__bf16)t3.z; ap.e[15] = (__bf16)t3.w;
        }

        // O += P * V : 4 WMMAs across head_dim.
#pragma unroll
        for (int dt = 0; dt < 4; dt++) {
            Frag bv;
            load_fragB(bv, Vs + dt * 16, PV);
            o[dt] = wmma_bf16(ap, bv, o[dt]);
        }
    }

    // epilogue: normalize and store bf16 context (token-major).
    {
        int cn = lane & 15, hf = lane >> 4;
#pragma unroll
        for (int dt = 0; dt < 4; dt++) {
#pragma unroll
            for (int r = 0; r < 8; r++) {
                float val = o[dt][r] / lrow[r];
                size_t row = (size_t)(b * 2048 + q0 + hf * 8 + r);
                ctx[row * 1024 + h * 64 + dt * 16 + cn] = (__bf16)val;
            }
        }
    }
}

// ---------------------------------------------------------------------------
extern "C" void kernel_launch(void* const* d_in, const int* in_sizes, int n_in,
                              void* d_out, int out_size, void* d_ws, size_t ws_size,
                              hipStream_t stream) {
    const float* z  = (const float*)d_in[0];
    const float* wq = (const float*)d_in[1];
    const float* wk = (const float*)d_in[2];
    const float* wv = (const float*)d_in[3];
    const float* wo = (const float*)d_in[4];
    const float* bo = (const float*)d_in[5];
    float* out = (float*)d_out;

    char* ws = (char*)d_ws;
    const size_t MB = 1024ull * 1024ull;
    __bf16* zb  = (__bf16*)(ws + 0);        // 16 MB  z bf16 [8192][1024]
    __bf16* wqb = (__bf16*)(ws + 16 * MB);  // 2 MB
    __bf16* wkb = (__bf16*)(ws + 18 * MB);  // 2 MB
    __bf16* wvb = (__bf16*)(ws + 20 * MB);  // 2 MB
    __bf16* wob = (__bf16*)(ws + 22 * MB);  // 2 MB
    __bf16* Qb  = (__bf16*)(ws + 24 * MB);  // 16 MB  Q  [8192][1024]
    __bf16* Ktb = (__bf16*)(ws + 40 * MB);  // 16 MB  K^T [1024][8192]
    __bf16* Vb  = (__bf16*)(ws + 56 * MB);  // 16 MB  V  [8192][1024]
    __bf16* ctx = (__bf16*)(ws + 72 * MB);  // 16 MB  attn output [8192][1024]

    // bf16 conversions
    cvt_kernel<<<8192, 256, 0, stream>>>(z, zb, 8192 * 1024 / 4);
    cvt_kernel<<<1024, 256, 0, stream>>>(wq, wqb, 1024 * 1024 / 4);
    cvt_kernel<<<1024, 256, 0, stream>>>(wk, wkb, 1024 * 1024 / 4);
    cvt_kernel<<<1024, 256, 0, stream>>>(wv, wvb, 1024 * 1024 / 4);
    cvt_kernel<<<1024, 256, 0, stream>>>(wo, wob, 1024 * 1024 / 4);

    // projections: (M/64)*(N/32)/4 waves-per-block = 1024 blocks
    gemm_kernel<<<1024, 128, 0, stream>>>(zb, wqb, Qb, nullptr, 8192, 1024, 1024, 0, 0);
    gemm_kernel<<<1024, 128, 0, stream>>>(zb, wkb, Ktb, nullptr, 8192, 1024, 1024, 1, 8192);
    gemm_kernel<<<1024, 128, 0, stream>>>(zb, wvb, Vb, nullptr, 8192, 1024, 1024, 0, 0);

    // flash attention: 32 query blocks x 64 (batch,head)
    attn_kernel<<<dim3(32, 64), 128, 0, stream>>>(Qb, Ktb, Vb, ctx);

    // output projection with bias, f32 out
    gemm_kernel<<<1024, 128, 0, stream>>>(ctx, wob, out, bo, 8192, 1024, 1024, 2, 0);
}
